// TaylorCNNv8_73091753443741
// MI455X (gfx1250) — matrix operand order repaired and checked by reading, not compiled, not run on hardware
//
#include <hip/hip_runtime.h>
#include <math.h>

typedef __attribute__((ext_vector_type(16))) _Float16 v16h;
typedef __attribute__((ext_vector_type(8)))  _Float16 v8h;
typedef __attribute__((ext_vector_type(4)))  _Float16 v4h;
typedef __attribute__((ext_vector_type(8)))  float    v8f;

#define L_LEN      2001
#define NTILE      126        // ceil(2001/16)
#define CH_POS     2048       // positions incl. margins (4 halfs each)
#define MID_POS    2048       // positions incl. margins (8 halfs each)
#define LP         4          // chbuf left margin (conv1 reads pos-3)
#define MLP        2          // midbuf left margin (conv2 reads pos-2)
#define NWAVE      4
#define BLOCK      128

// B fragment for conv1: K = h + 16*hi contiguous halfs, 8B aligned -> ds_load_b64 x4
static __device__ __forceinline__ v16h load_b1(const _Float16* chb, int p0, int nlo, int hi) {
    union { v16h v; uint2 q[4]; } u;
    const uint2* s = (const uint2*)&chb[((LP + p0 + nlo - 3) << 2) + (hi << 4)];
    u.q[0] = s[0]; u.q[1] = s[1]; u.q[2] = s[2]; u.q[3] = s[3];
    return u.v;
}

// B fragments for conv2: 16B aligned -> ds_load_b128 x2 each
static __device__ __forceinline__ v16h load_b2(const _Float16* midb, int base2, int koff) {
    union { v16h v; uint4 o[2]; } u;
    const uint4* s = (const uint4*)&midb[base2 + koff];
    u.o[0] = s[0]; u.o[1] = s[1];
    return u.v;
}

__global__ __launch_bounds__(BLOCK)
void taylor_cnn_v8_kernel(
    const float* __restrict__ phase,
    const float* __restrict__ pflux,
    const float* __restrict__ sflux,
    const float* __restrict__ oediff,
    const float* __restrict__ Aparm,    // [1]
    const float* __restrict__ Bparm,    // [1]
    const float* __restrict__ t0parm,   // [1]
    const float* __restrict__ bn_gamma, // [4]
    const float* __restrict__ bn_beta,  // [4]
    const float* __restrict__ bn_mean,  // [4]
    const float* __restrict__ bn_var,   // [4]
    const float* __restrict__ w1,       // (8,4,7)
    const float* __restrict__ b1,       // (8)
    const float* __restrict__ w2,       // (16,8,5)
    const float* __restrict__ b2,       // (16)
    const float* __restrict__ fcw,      // (21)
    const float* __restrict__ fcb,      // (1)
    float* __restrict__ out)            // (B)
{
    // channel-interleaved activations: chbuf[(LP+pos)*4 + c], midbuf[(MLP+pos)*8 + c]
    __shared__ _Float16 chbuf[4 * CH_POS];     // 16 KB
    __shared__ _Float16 midbuf[8 * MID_POS];   // 32 KB
    __shared__ float    red[NWAVE][32][8];     // 4 KB
    __shared__ float    sacc[3];               // ing_sum, fb_sum, auc_interp_sum
    __shared__ float    cnn[16];

    const int b    = blockIdx.x;
    const int tid  = threadIdx.x;
    const int wave = tid >> 5;
    const int lane = tid & 31;
    const int hi   = (lane >> 4) & 1;   // lane half: selects K/M block per ISA layout
    const int nlo  = lane & 15;

    // ---- zero LDS with wide stores (margins = implicit conv zero-padding) ----
    {
        v8h z = {};
        v8h* zc = (v8h*)chbuf;   // 1024 chunks
        v8h* zm = (v8h*)midbuf;  // 2048 chunks
        for (int i = tid; i < 1024; i += BLOCK) zc[i] = z;
        for (int i = tid; i < 2048; i += BLOCK) zm[i] = z;
    }
    if (tid < 3) sacc[tid] = 0.f;
    __syncthreads();

    const float A   = Aparm[0];
    const float Bp  = Bparm[0];
    const float t0  = t0parm[0];
    const float A_abs = fabsf(A) + 1e-8f;

    float bninv[4], bnsh[4];
    #pragma unroll
    for (int c = 0; c < 4; ++c) {
        bninv[c] = bn_gamma[c] / sqrtf(bn_var[c] + 1e-5f);
        bnsh[c]  = bn_beta[c] - bn_mean[c] * bninv[c];
    }

    const size_t rowoff = (size_t)b * (size_t)L_LEN;
    const float* prow = pflux + rowoff;

    // ---- phase 0: gate + BN -> interleaved LDS (b64 stores); fused shape sums ----
    float ing_part = 0.f, fb_part = 0.f;
    for (int i = tid; i < L_LEN; i += BLOCK) {
        float ph  = phase[rowoff + i];
        float dph = ph - t0;
        float g   = fmaxf(1.f - Bp * dph * dph, 0.f);
        g = -A * g;
        float pf = prow[i];
        float sf = sflux[rowoff + i];
        float oe = oediff[rowoff + i];
        v4h cv;
        cv[0] = (_Float16)(g  * bninv[0] + bnsh[0]);
        cv[1] = (_Float16)(pf * bninv[1] + bnsh[1]);
        cv[2] = (_Float16)(sf * bninv[2] + bnsh[2]);
        cv[3] = (_Float16)(oe * bninv[3] + bnsh[3]);
        *(v4h*)&chbuf[(LP + i) << 2] = cv;   // 8B aligned -> ds_store_b64

        float d  = fmaxf(-pf, 0.f) / A_abs;
        float s1 = 1.f / (1.f + __expf(-50.f * (d - 0.1f)));
        float s2 = 1.f / (1.f + __expf(-50.f * (d - 0.8f)));
        float ing = s1 - s2;
        float tot = 1.f - s1 + ing + s2 + 1e-8f;
        ing_part += ing / tot;
        fb_part  += s2  / tot;
    }
    atomicAdd(&sacc[0], ing_part);
    atomicAdd(&sacc[1], fb_part);

    float auc_part = 0.f;
    for (int i = tid; i < 200; i += BLOCK) {
        float pos = (float)i * (2000.0f / 199.0f);
        int i0 = (int)pos;
        int i1 = i0 + 1; if (i1 > 2000) i1 = 2000;
        float w  = pos - (float)i0;
        float d0 = fmaxf(-prow[i0], 0.f);
        float d1 = fmaxf(-prow[i1], 0.f);
        auc_part += d0 * (1.f - w) + d1 * w;
    }
    atomicAdd(&sacc[2], auc_part);
    __syncthreads();

    // ================= conv1 (4->8, k=7) : WMMA f16, K' = t*4+c (28 -> 32) =======
    // A layout (16-bit A 16x32): half h -> K = h + 16*hi (matches contiguous B loads)
    v16h a1;
    #pragma unroll
    for (int h = 0; h < 16; ++h) {
        int g = h >> 1, r = h & 1;
        int k = h + 16 * hi;  // K' index
        float wv = 0.f;
        if (nlo < 8 && k < 28) wv = w1[nlo * 28 + (k & 3) * 7 + (k >> 2)];
        a1[h] = (_Float16)wv;
    }
    float bb1[8];
    #pragma unroll
    for (int i = 0; i < 8; ++i) bb1[i] = b1[i];

    // D: vgpr i, lanes 0-15 -> channel i, N = p0+nlo. Store interleaved b128.
    #define CONV1_STORE(TILE, ACC)                                              \
        {                                                                       \
            const int pos_ = ((TILE) << 4) + nlo;                               \
            if (hi == 0 && pos_ < L_LEN) {                                      \
                v8h mv;                                                         \
                _Pragma("unroll")                                               \
                for (int i_ = 0; i_ < 8; ++i_)                                  \
                    mv[i_] = (_Float16)fmaxf((ACC)[i_] + bb1[i_], 0.f);         \
                *(v8h*)&midbuf[(MLP + pos_) << 3] = mv;                         \
            }                                                                   \
        }

    {
        int tile = wave;
        // 2x software-pipelined: both tiles' B loads issued before either WMMA
        for (; tile + NWAVE < NTILE; tile += 2 * NWAVE) {
            v16h u0 = load_b1(chbuf, tile << 4, nlo, hi);
            v16h u1 = load_b1(chbuf, (tile + NWAVE) << 4, nlo, hi);
            v8f acc0 = {}, acc1 = {};
            acc0 = __builtin_amdgcn_wmma_f32_16x16x32_f16(
                       false, a1, false, u0, (short)0, acc0, false, false);
            acc1 = __builtin_amdgcn_wmma_f32_16x16x32_f16(
                       false, a1, false, u1, (short)0, acc1, false, false);
            CONV1_STORE(tile, acc0);
            CONV1_STORE(tile + NWAVE, acc1);
        }
        for (; tile < NTILE; tile += NWAVE) {
            v16h u0 = load_b1(chbuf, tile << 4, nlo, hi);
            v8f acc0 = {};
            acc0 = __builtin_amdgcn_wmma_f32_16x16x32_f16(
                       false, a1, false, u0, (short)0, acc0, false, false);
            CONV1_STORE(tile, acc0);
        }
    }
    __syncthreads();

    // ================= conv2 (8->16, k=5) : K' = t*8+c (40 -> 2x32) ==============
    v16h a2lo, a2hi;
    #pragma unroll
    for (int h = 0; h < 16; ++h) {
        int g = h >> 1, r = h & 1;
        int k  = h + 16 * hi;  // 0..31, always < 40
        int k2 = k + 32;                                     // 32..63
        a2lo[h] = (_Float16)w2[nlo * 40 + (k & 7) * 5 + (k >> 3)];
        a2hi[h] = (_Float16)((k2 < 40) ? w2[nlo * 40 + (k2 & 7) * 5 + (k2 >> 3)] : 0.f);
    }
    float bias2[8];
    #pragma unroll
    for (int i = 0; i < 8; ++i) bias2[i] = b2[i + 8 * hi];

    float sums[8];
    #pragma unroll
    for (int i = 0; i < 8; ++i) sums[i] = 0.f;

    #define CONV2_ACCUM(TILE, ACC)                                              \
        {                                                                       \
            if (((TILE) << 4) + nlo < L_LEN) {                                  \
                _Pragma("unroll")                                               \
                for (int i_ = 0; i_ < 8; ++i_)                                  \
                    sums[i_] += fmaxf((ACC)[i_] + bias2[i_], 0.f);              \
            }                                                                   \
        }

    {
        int tile = wave;
        for (; tile + NWAVE < NTILE; tile += 2 * NWAVE) {
            const int base0 = ((MLP + (tile << 4) + nlo - 2) << 3) + (hi << 4);
            const int base1 = ((MLP + ((tile + NWAVE) << 4) + nlo - 2) << 3) + (hi << 4);
            v16h blo0 = load_b2(midbuf, base0, 0);
            v16h bhi0 = load_b2(midbuf, base0, 32);
            v16h blo1 = load_b2(midbuf, base1, 0);
            v16h bhi1 = load_b2(midbuf, base1, 32);
            v8f acc0 = {}, acc1 = {};
            acc0 = __builtin_amdgcn_wmma_f32_16x16x32_f16(
                       false, a2lo, false, blo0, (short)0, acc0, false, false);
            acc0 = __builtin_amdgcn_wmma_f32_16x16x32_f16(
                       false, a2hi, false, bhi0, (short)0, acc0, false, false);
            acc1 = __builtin_amdgcn_wmma_f32_16x16x32_f16(
                       false, a2lo, false, blo1, (short)0, acc1, false, false);
            acc1 = __builtin_amdgcn_wmma_f32_16x16x32_f16(
                       false, a2hi, false, bhi1, (short)0, acc1, false, false);
            CONV2_ACCUM(tile, acc0);
            CONV2_ACCUM(tile + NWAVE, acc1);
        }
        for (; tile < NTILE; tile += NWAVE) {
            const int base0 = ((MLP + (tile << 4) + nlo - 2) << 3) + (hi << 4);
            v16h blo0 = load_b2(midbuf, base0, 0);
            v16h bhi0 = load_b2(midbuf, base0, 32);
            v8f acc0 = {};
            acc0 = __builtin_amdgcn_wmma_f32_16x16x32_f16(
                       false, a2lo, false, blo0, (short)0, acc0, false, false);
            acc0 = __builtin_amdgcn_wmma_f32_16x16x32_f16(
                       false, a2hi, false, bhi0, (short)0, acc0, false, false);
            CONV2_ACCUM(tile, acc0);
        }
    }
    #pragma unroll
    for (int i = 0; i < 8; ++i) red[wave][lane][i] = sums[i];
    __syncthreads();

    // ---- reduce: channel m = i + 8*hi; lanes 0-15 hold hi=0, 16-31 hold hi=1 ----
    if (tid < 16) {
        int i = tid & 7;
        int base = (tid >> 3) << 4; // 0 or 16
        float s = 0.f;
        for (int w = 0; w < NWAVE; ++w)
            #pragma unroll
            for (int l = 0; l < 16; ++l)
                s += red[w][base + l][i];
        cnn[tid] = s / (float)L_LEN;
    }
    __syncthreads();

    // ---- FC(21->1) + sigmoid ----
    if (tid == 0) {
        float ing_mean = sacc[0] / (float)L_LEN;
        float fb_mean  = sacc[1] / (float)L_LEN;
        float t14      = ing_mean + fb_mean + 1e-8f;
        float t12_t14  = ing_mean / t14;
        float rs0   = fmaxf(-prow[0],    0.f);
        float rs199 = fmaxf(-prow[2000], 0.f);
        float dx = 2.f / 199.f;
        float auc_raw  = dx * (sacc[2] - 0.5f * (rs0 + rs199));
        float auc_norm = auc_raw / A_abs;

        float z = fcb[0];
        #pragma unroll
        for (int j = 0; j < 16; ++j) z += fcw[j] * cnn[j];
        z += fcw[16] * Bp;
        z += fcw[17] * auc_raw;
        z += fcw[18] * auc_norm;
        z += fcw[19] * t12_t14;
        z += fcw[20] * fb_mean;
        out[b] = 1.f / (1.f + __expf(-z));
    }
}

extern "C" void kernel_launch(void* const* d_in, const int* in_sizes, int n_in,
                              void* d_out, int out_size, void* d_ws, size_t ws_size,
                              hipStream_t stream) {
    (void)n_in; (void)d_ws; (void)ws_size; (void)out_size;
    const float* phase    = (const float*)d_in[0];
    const float* pflux    = (const float*)d_in[1];
    const float* sflux    = (const float*)d_in[2];
    const float* oediff   = (const float*)d_in[3];
    const float* Aparm    = (const float*)d_in[4];
    const float* Bparm    = (const float*)d_in[5];
    const float* t0parm   = (const float*)d_in[6];
    const float* bn_gamma = (const float*)d_in[7];
    const float* bn_beta  = (const float*)d_in[8];
    const float* bn_mean  = (const float*)d_in[9];
    const float* bn_var   = (const float*)d_in[10];
    const float* w1       = (const float*)d_in[11];
    const float* b1       = (const float*)d_in[12];
    const float* w2       = (const float*)d_in[13];
    const float* b2       = (const float*)d_in[14];
    const float* fcw      = (const float*)d_in[15];
    const float* fcb      = (const float*)d_in[16];
    float* out            = (float*)d_out;

    const int B = in_sizes[0] / L_LEN;  // 8192
    taylor_cnn_v8_kernel<<<B, BLOCK, 0, stream>>>(
        phase, pflux, sflux, oediff, Aparm, Bparm, t0parm,
        bn_gamma, bn_beta, bn_mean, bn_var,
        w1, b1, w2, b2, fcw, fcb, out);
}
